// ActTransNet_31284541784432
// MI455X (gfx1250) — matrix operand order repaired and checked by reading, not compiled, not run on hardware
//
#include <hip/hip_runtime.h>
#include <hip/hip_bf16.h>
#include <stdint.h>

#define INPUT_DIM 2048
#define DIM 1024
#define N_ACTIONS 64
#define BATCH 1024
#define TSTEPS 16
#define ASTR 72   // LDS row stride in ushorts for 64-wide K tiles (144B rows, 16B aligned)

typedef __attribute__((ext_vector_type(16))) __bf16 v16bf;
typedef __attribute__((ext_vector_type(8)))  float  v8f;

union Frag { uint4 q[2]; v16bf v; };

__device__ __forceinline__ unsigned short f2bf(float f) {
    unsigned int u = __float_as_uint(f);
    u += 0x7FFFu + ((u >> 16) & 1u);          // round-to-nearest-even
    return (unsigned short)(u >> 16);
}
__device__ __forceinline__ unsigned int pack2bf(float lo, float hi) {
    return (unsigned int)f2bf(lo) | ((unsigned int)f2bf(hi) << 16);
}

// Raw 32-bit LDS byte offset of a __shared__ object (generic -> AS(3) -> int).
__device__ __forceinline__ unsigned lds_off_of(const void* p) {
    return (unsigned)(unsigned long long)
        (__attribute__((address_space(3))) const char*)p;
}

// CDNA5 async global->LDS copy, 16B per lane, tracked by ASYNCcnt.
#define ASYNC_LDS_B128(ldsOff, gptr)                                   \
    asm volatile("global_load_async_to_lds_b128 %0, %1, off"           \
                 :: "v"(ldsOff), "v"(gptr) : "memory")
#define WAIT_ASYNC0() asm volatile("s_wait_asynccnt 0x0" ::: "memory")

// ---------------------------------------------------------------------------
// Kernel 1: mean over T (=16) + fp32 -> bf16 convert.  z=0: precondition, z=1: effect
// float4 streaming loads, uint2 packed bf16 stores.
// ---------------------------------------------------------------------------
__global__ __launch_bounds__(256) void mean_cvt(
    const float* __restrict__ pre, const float* __restrict__ eff,
    unsigned short* __restrict__ outP, unsigned short* __restrict__ outE)
{
    const int z = blockIdx.z;
    const float* in = z ? eff : pre;
    unsigned short* out = z ? outE : outP;
    const int b = blockIdx.y;
    const int k4 = blockIdx.x * 256 + threadIdx.x;       // float4 index in row
    const float4* base = reinterpret_cast<const float4*>(
        in + (size_t)b * TSTEPS * INPUT_DIM) + k4;
    float4 s = make_float4(0.f, 0.f, 0.f, 0.f);
    #pragma unroll
    for (int t = 0; t < TSTEPS; ++t) {
        float4 v = base[(size_t)t * (INPUT_DIM / 4)];
        s.x += v.x; s.y += v.y; s.z += v.z; s.w += v.w;
    }
    const float sc = 1.0f / TSTEPS;
    uint2 o;
    o.x = pack2bf(s.x * sc, s.y * sc);
    o.y = pack2bf(s.z * sc, s.w * sc);
    reinterpret_cast<uint2*>(out + (size_t)b * INPUT_DIM)[k4] = o;
}

// ---------------------------------------------------------------------------
// Kernel 2: group batch indices by action (perm[a][slot], counts[a])
// ---------------------------------------------------------------------------
__global__ void build_perm(const int* __restrict__ action,
                           int* __restrict__ perm, int* __restrict__ counts)
{
    const int a = threadIdx.x;               // 64 threads
    int c = 0;
    for (int b = 0; b < BATCH; ++b)
        if (action[b] == a) perm[a * BATCH + (c++)] = b;
    counts[a] = c;
}

// ---------------------------------------------------------------------------
// Kernel 3: embed GEMM  (M=1024, N=1024, K=2048), z=0: p path (bf16 out), z=1: e path (f32 out+bias)
// Workgroup tile 64x64, 8 waves as 4(M) x 2(N), K-step 64 -> 4 WMMAs/wave/step.
// A tiles (bf16) staged via async global->LDS b128; B converted fp32->bf16 in flight.
// ---------------------------------------------------------------------------
__global__ __launch_bounds__(256) void embed_gemm(
    const unsigned short* __restrict__ Ap, const unsigned short* __restrict__ Ae,
    const float* __restrict__ Wp, const float* __restrict__ We,
    const float* __restrict__ bp, const float* __restrict__ be,
    unsigned short* __restrict__ p_embed_bf, float* __restrict__ e_out)
{
    const int z = blockIdx.z;
    const unsigned short* A = z ? Ae : Ap;
    const float* W = z ? We : Wp;
    const float* bias = z ? be : bp;

    __shared__ unsigned short Ash[64 * ASTR];   // [m][k] bf16, 64x64 tile
    __shared__ unsigned short Bsh[64 * ASTR];   // [n][k] bf16, 64x64 tile

    const int tid  = threadIdx.x;
    const int lane = tid & 31;
    const int wave = tid >> 5;
    const int wm = wave >> 1;                   // 0..3
    const int wn = wave & 1;                    // 0..1
    const int m0 = blockIdx.y * 64;
    const int n0 = blockIdx.x * 64;

    v8f acc0 = {}; v8f acc1 = {};

    for (int kk = 0; kk < INPUT_DIM; kk += 64) {
        // A: 64 rows x 64 bf16 = 512 x 16B chunks, async copy, 2 per thread
        #pragma unroll
        for (int i = 0; i < 2; ++i) {
            int u = tid + i * 256;
            int r = u >> 3, c = u & 7;                       // 8 chunks per row
            const unsigned short* gsrc = A + (size_t)(m0 + r) * INPUT_DIM + kk + c * 8;
            ASYNC_LDS_B128(lds_off_of(&Ash[r * ASTR + c * 8]), gsrc);
        }
        // B: 64 rows x 64 f32 -> bf16, 4 float4 per thread
        #pragma unroll
        for (int i = 0; i < 4; ++i) {
            int u = tid + i * 256;
            int r = u >> 4, c = u & 15;                      // 16 float4 per row
            const float4* src =
                reinterpret_cast<const float4*>(W + (size_t)(n0 + r) * INPUT_DIM + kk);
            float4 v = src[c];
            unsigned int* dst = reinterpret_cast<unsigned int*>(&Bsh[r * ASTR + c * 4]);
            dst[0] = pack2bf(v.x, v.y);
            dst[1] = pack2bf(v.z, v.w);
        }
        WAIT_ASYNC0();
        __syncthreads();

        const int mrow = wm * 16 + (lane & 15);
        const int ka   = (lane >> 4) * 8;
        const int kb   = (lane >> 4) * 8;
        const int nrow0 = wn * 32 + (lane & 15);
        const int nrow1 = nrow0 + 16;
        #pragma unroll
        for (int kc = 0; kc < 64; kc += 32) {
            Frag a, b0, b1;
            a.q[0]  = *reinterpret_cast<const uint4*>(&Ash[mrow * ASTR + kc + ka]);
            a.q[1]  = *reinterpret_cast<const uint4*>(&Ash[mrow * ASTR + kc + ka + 16]);
            b0.q[0] = *reinterpret_cast<const uint4*>(&Bsh[nrow0 * ASTR + kc + kb]);
            b0.q[1] = *reinterpret_cast<const uint4*>(&Bsh[nrow0 * ASTR + kc + kb + 16]);
            b1.q[0] = *reinterpret_cast<const uint4*>(&Bsh[nrow1 * ASTR + kc + kb]);
            b1.q[1] = *reinterpret_cast<const uint4*>(&Bsh[nrow1 * ASTR + kc + kb + 16]);
            acc0 = __builtin_amdgcn_wmma_f32_16x16x32_bf16(false, a.v, false, b0.v,
                                                           (short)0, acc0, false, false);
            acc1 = __builtin_amdgcn_wmma_f32_16x16x32_bf16(false, a.v, false, b1.v,
                                                           (short)0, acc1, false, false);
        }
        __syncthreads();
    }

    const int mg = m0 + wm * 16 + (lane >> 4) * 8;   // D: VGPR r <-> M = (l>>4)*8 + r
    const int ng = n0 + wn * 32 + (lane & 15);
    #pragma unroll
    for (int r = 0; r < 8; ++r) {
        const int m = mg + r;
        const float v0 = acc0[r] + bias[ng];
        const float v1 = acc1[r] + bias[ng + 16];
        if (z) {
            e_out[(size_t)m * DIM + ng]      = v0;
            e_out[(size_t)m * DIM + ng + 16] = v1;
        } else {
            p_embed_bf[(size_t)m * DIM + ng]      = f2bf(v0);
            p_embed_bf[(size_t)m * DIM + ng + 16] = f2bf(v1);
        }
    }
}

// ---------------------------------------------------------------------------
// Kernel 4: per-action transform.  For action a: out[rows] = P_a @ W_trans[a]^T
// Grid: x = N-chunk (8 x 128 cols), y = action. 8 waves, one 16-col tile each.
// K-step 64 -> 2 WMMAs/wave/step. Gathered A rows staged async; W_trans fp32->bf16 in flight.
// ---------------------------------------------------------------------------
__global__ __launch_bounds__(256) void act_transform(
    const unsigned short* __restrict__ p_embed_bf,
    const float* __restrict__ W_trans,
    const int* __restrict__ perm, const int* __restrict__ counts,
    float* __restrict__ outP)
{
    const int a = blockIdx.y;
    const int cnt = counts[a];
    if (cnt == 0) return;
    const int mtiles = (cnt + 15) >> 4;
    const int nbase = blockIdx.x * 128;

    __shared__ unsigned short Ash[16 * ASTR];    // gathered p_embed rows [m][k], 16x64
    __shared__ unsigned short Bsh[128 * ASTR];   // W_trans[a] rows [n][k], 128x64
    __shared__ int rowIdx[16];

    const int tid  = threadIdx.x;
    const int lane = tid & 31;
    const int wave = tid >> 5;
    const float* Wa = W_trans + (size_t)a * DIM * DIM;

    for (int mt = 0; mt < mtiles; ++mt) {
        __syncthreads();
        if (tid < 16) {
            const int slot = mt * 16 + tid;
            rowIdx[tid] = (slot < cnt) ? perm[a * BATCH + slot] : 0;
        }
        __syncthreads();

        v8f acc = {};
        for (int kk = 0; kk < DIM; kk += 64) {
            // A: 16 gathered rows x 64 bf16 = 128 x 16B chunks, async copy
            if (tid < 128) {
                const int r = tid >> 3, c = tid & 7;
                const unsigned short* gsrc =
                    p_embed_bf + (size_t)rowIdx[r] * DIM + kk + c * 8;
                ASYNC_LDS_B128(lds_off_of(&Ash[r * ASTR + c * 8]), gsrc);
            }
            // B: 128 rows x 64 f32 -> bf16, 8 float4 per thread
            #pragma unroll
            for (int i = 0; i < 8; ++i) {
                int u = tid + i * 256;
                int r = u >> 4, c = u & 15;
                const float4* src =
                    reinterpret_cast<const float4*>(Wa + (size_t)(nbase + r) * DIM + kk);
                float4 v = src[c];
                unsigned int* dst = reinterpret_cast<unsigned int*>(&Bsh[r * ASTR + c * 4]);
                dst[0] = pack2bf(v.x, v.y);
                dst[1] = pack2bf(v.z, v.w);
            }
            WAIT_ASYNC0();
            __syncthreads();

            const int mrow = lane & 15;
            const int ka   = (lane >> 4) * 8;
            const int nrow = wave * 16 + (lane & 15);
            const int kb   = (lane >> 4) * 8;
            #pragma unroll
            for (int kc = 0; kc < 64; kc += 32) {
                Frag af, bf;
                af.q[0] = *reinterpret_cast<const uint4*>(&Ash[mrow * ASTR + kc + ka]);
                af.q[1] = *reinterpret_cast<const uint4*>(&Ash[mrow * ASTR + kc + ka + 16]);
                bf.q[0] = *reinterpret_cast<const uint4*>(&Bsh[nrow * ASTR + kc + kb]);
                bf.q[1] = *reinterpret_cast<const uint4*>(&Bsh[nrow * ASTR + kc + kb + 16]);
                acc = __builtin_amdgcn_wmma_f32_16x16x32_bf16(false, af.v, false, bf.v,
                                                              (short)0, acc, false, false);
            }
            __syncthreads();
        }

        const int ncol = nbase + wave * 16 + (lane & 15);
        #pragma unroll
        for (int r = 0; r < 8; ++r) {
            const int mloc = (lane >> 4) * 8 + r;
            const int slot = mt * 16 + mloc;
            if (slot < cnt) {
                const int b = rowIdx[mloc];
                outP[(size_t)b * DIM + ncol] = acc[r];
            }
        }
    }
}

// ---------------------------------------------------------------------------
extern "C" void kernel_launch(void* const* d_in, const int* in_sizes, int n_in,
                              void* d_out, int out_size, void* d_ws, size_t ws_size,
                              hipStream_t stream) {
    const float* pre    = (const float*)d_in[0];
    const float* eff    = (const float*)d_in[1];
    const int*   action = (const int*)  d_in[2];
    const float* Wp     = (const float*)d_in[3];
    const float* bp     = (const float*)d_in[4];
    const float* We     = (const float*)d_in[5];
    const float* be     = (const float*)d_in[6];
    const float* Wt     = (const float*)d_in[7];

    char* ws = (char*)d_ws;
    unsigned short* pavg   = (unsigned short*)(ws);                    // 4 MB
    unsigned short* eavg   = (unsigned short*)(ws + (4u  << 20));      // 4 MB
    unsigned short* pemb   = (unsigned short*)(ws + (8u  << 20));      // 2 MB
    int*            perm   = (int*)           (ws + (10u << 20));      // 256 KB
    int*            counts = (int*)           (ws + (10u << 20) + N_ACTIONS * BATCH * 4);

    float* outP = (float*)d_out;                          // p_transformed: (B,1,DIM,1) flat
    float* outE = (float*)d_out + (size_t)BATCH * DIM;    // e_embed

    mean_cvt <<<dim3(INPUT_DIM / 1024, BATCH, 2), 256, 0, stream>>>(pre, eff, pavg, eavg);
    build_perm<<<1, N_ACTIONS, 0, stream>>>(action, perm, counts);
    embed_gemm<<<dim3(DIM / 64, BATCH / 64, 2), 256, 0, stream>>>(
        pavg, eavg, Wp, We, bp, be, pemb, outE);
    act_transform<<<dim3(DIM / 128, N_ACTIONS), 256, 0, stream>>>(
        pemb, Wt, perm, counts, outP);
}
